// KANLayer_4234837754132
// MI455X (gfx1250) — compile-verified, matrix operand order checked by reading, not run on hardware
//
#include <hip/hip_runtime.h>
#include <hip/hip_bf16.h>

typedef _Float16 half_t;
typedef __attribute__((ext_vector_type(16))) _Float16 v16h;
typedef __attribute__((ext_vector_type(8)))  _Float16 v8h;
typedef __attribute__((ext_vector_type(8)))  float    v8f;
typedef int v4i_t __attribute__((vector_size(16)));

#define B_SZ   8192
#define IN_SZ  1024
#define OUT_SZ 1024
#define KSPL   8
#define K1     IN_SZ            // base GEMM K = 1024
#define K2     (IN_SZ * KSPL)   // spline GEMM K = 8192
#define BLK_M  128
#define BLK_N  128
#define BLK_K  32
#define NKT    ((K1 + K2) / BLK_K)  // 288 k-tiles total
#define NKT1   (K1 / BLK_K)         // 32 tiles in base phase

#if defined(__has_builtin)
#if __has_builtin(__builtin_amdgcn_global_load_async_to_lds_b128) && \
    __has_builtin(__builtin_amdgcn_s_wait_asynccnt)
#define USE_ASYNC_LDS 1
#endif
#endif

#ifdef USE_ASYNC_LDS
__device__ __forceinline__ void async_cp16(const half_t* g, half_t* l) {
  __builtin_amdgcn_global_load_async_to_lds_b128(
      (__attribute__((address_space(1))) v4i_t*)(unsigned long long)g,
      (__attribute__((address_space(3))) v4i_t*)(unsigned long long)l,
      0, 0);
}
#endif

// ---------------------------------------------------------------------------
// Kernel 1: GELU(x) -> Ag (f16), cubic B-spline bases of x -> As (f16)
// ---------------------------------------------------------------------------
__global__ __launch_bounds__(256) void kan_prep(
    const float* __restrict__ x, const float* __restrict__ grid,
    half_t* __restrict__ Ag, half_t* __restrict__ As)
{
  size_t idx = (size_t)blockIdx.x * 256 + threadIdx.x;   // over B*IN, exact
  int i = (int)(idx % IN_SZ);
  float xv = x[idx];

  float ge = 0.5f * xv * (1.0f + erff(xv * 0.70710678118654752f));
  Ag[idx] = (half_t)ge;

  float g[12];
#pragma unroll
  for (int j = 0; j < 12; ++j) g[j] = grid[(size_t)i * 12 + j];

  float bas[11];
#pragma unroll
  for (int j = 0; j < 11; ++j)
    bas[j] = (xv >= g[j] && xv < g[j + 1]) ? 1.0f : 0.0f;

#pragma unroll
  for (int k = 1; k <= 3; ++k) {
#pragma unroll
    for (int j = 0; j + k < 11; ++j) {
      float dr = g[j + k] - g[j];         dr = (dr == 0.0f) ? 1.0f : dr;
      float dd = g[j + k + 1] - g[j + 1]; dd = (dd == 0.0f) ? 1.0f : dd;
      bas[j] = (xv - g[j]) / dr * bas[j] + (g[j + k + 1] - xv) / dd * bas[j + 1];
    }
  }

  size_t b = idx / IN_SZ;
  half_t* dst = As + b * (size_t)K2 + (size_t)i * KSPL;
#pragma unroll
  for (int s = 0; s < KSPL; ++s) dst[s] = (half_t)bas[s];
}

// ---------------------------------------------------------------------------
// Kernel 2: weights fp32 -> f16 (layouts are already [N][K] row-major)
// ---------------------------------------------------------------------------
__global__ __launch_bounds__(256) void kan_pack(
    const float* __restrict__ bw, const float* __restrict__ sw,
    half_t* __restrict__ Wbh, half_t* __restrict__ Wsh)
{
  size_t idx = (size_t)blockIdx.x * 256 + threadIdx.x;
  const size_t n1 = (size_t)OUT_SZ * K1;
  if (idx < n1) Wbh[idx] = (half_t)bw[idx];
  else          Wsh[idx - n1] = (half_t)sw[idx - n1];
}

// ---------------------------------------------------------------------------
// Kernel 3: fused dual-phase WMMA GEMM  Y = Ag*Wb^T + As*Ws^T  (f32 accum)
// ---------------------------------------------------------------------------
__device__ __forceinline__ void load_tiles(
    int kt, int m0, int n0,
    const half_t* __restrict__ Ag, const half_t* __restrict__ As,
    const half_t* __restrict__ Wbh, const half_t* __restrict__ Wsh,
    half_t* __restrict__ Atile, half_t* __restrict__ Btile, int tid)
{
  const half_t* pa; const half_t* pb; int lda, ldb, kk;
  if (kt < NKT1) {
    pa = Ag + (size_t)m0 * K1; lda = K1;
    pb = Wbh + (size_t)n0 * K1; ldb = K1;
    kk = kt * BLK_K;
  } else {
    pa = As + (size_t)m0 * K2; lda = K2;
    pb = Wsh + (size_t)n0 * K2; ldb = K2;
    kk = (kt - NKT1) * BLK_K;
  }
  // 128 rows x 32 halfs = 512 chunks of 16B per tile; 2 chunks/thread/tile
  for (int c = tid; c < (BLK_M * BLK_K) / 8; c += 256) {
    int r  = c >> 2;
    int kc = (c & 3) * 8;
    const half_t* gA = pa + (size_t)r * lda + kk + kc;
    const half_t* gB = pb + (size_t)r * ldb + kk + kc;
    half_t* lA = Atile + r * BLK_K + kc;
    half_t* lB = Btile + r * BLK_K + kc;
#ifdef USE_ASYNC_LDS
    async_cp16(gA, lA);
    async_cp16(gB, lB);
#else
    *(v8h*)lA = *(const v8h*)gA;
    *(v8h*)lB = *(const v8h*)gB;
#endif
    // prefetch the tile-after-next into L2
    __builtin_prefetch((const void*)(gA + 2 * BLK_K), 0, 1);
    __builtin_prefetch((const void*)(gB + 2 * BLK_K), 0, 1);
  }
}

__global__ __launch_bounds__(256) void kan_wmma_gemm(
    const half_t* __restrict__ Ag, const half_t* __restrict__ As,
    const half_t* __restrict__ Wbh, const half_t* __restrict__ Wsh,
    float* __restrict__ Y)
{
  __shared__ alignas(32) half_t At[2][BLK_M * BLK_K];
  __shared__ alignas(32) half_t Bt[2][BLK_N * BLK_K];

  const int tid  = threadIdx.x;
  const int m0   = blockIdx.y * BLK_M;
  const int n0   = blockIdx.x * BLK_N;
  const int lane = tid & 31;
  const int wave = tid >> 5;
  const int wm   = wave >> 1;        // 0..3 -> m offset wm*32
  const int wn   = wave & 1;         // 0..1 -> n offset wn*64
  const int lg   = lane & 15;
  const int kh   = (lane >> 4) ? 16 : 0;   // which 16-half K group this lane holds

  v8f acc[2][4];
#pragma unroll
  for (int mi = 0; mi < 2; ++mi)
#pragma unroll
    for (int ni = 0; ni < 4; ++ni) acc[mi][ni] = (v8f){};

  load_tiles(0, m0, n0, Ag, As, Wbh, Wsh, At[0], Bt[0], tid);
#ifdef USE_ASYNC_LDS
  __builtin_amdgcn_s_wait_asynccnt(0);
#endif
  __syncthreads();

  int buf = 0;
  for (int kt = 0; kt < NKT; ++kt) {
    if (kt + 1 < NKT)
      load_tiles(kt + 1, m0, n0, Ag, As, Wbh, Wsh, At[buf ^ 1], Bt[buf ^ 1], tid);

    const half_t* Ab = At[buf];
    const half_t* Bb = Bt[buf];
    v16h af[2], bf[4];
#pragma unroll
    for (int mi = 0; mi < 2; ++mi)
      af[mi] = *(const v16h*)(Ab + (wm * 32 + mi * 16 + lg) * BLK_K + kh);
#pragma unroll
    for (int ni = 0; ni < 4; ++ni)
      bf[ni] = *(const v16h*)(Bb + (wn * 64 + ni * 16 + lg) * BLK_K + kh);
#pragma unroll
    for (int mi = 0; mi < 2; ++mi)
#pragma unroll
      for (int ni = 0; ni < 4; ++ni)
        acc[mi][ni] = __builtin_amdgcn_wmma_f32_16x16x32_f16(
            false, af[mi], false, bf[ni], (short)0, acc[mi][ni], false, false);

#ifdef USE_ASYNC_LDS
    __builtin_amdgcn_s_wait_asynccnt(0);
#endif
    __syncthreads();
    buf ^= 1;
  }

  // Epilogue: C/D layout — VGPR r: lanes 0-15 -> M=m+r, lanes 16-31 -> M=m+8+r
  const int lh = lane >> 4;
#pragma unroll
  for (int mi = 0; mi < 2; ++mi) {
#pragma unroll
    for (int ni = 0; ni < 4; ++ni) {
      int mrow = m0 + wm * 32 + mi * 16 + lh * 8;
      int ncol = n0 + wn * 64 + ni * 16 + lg;
#pragma unroll
      for (int r = 0; r < 8; ++r)
        Y[(size_t)(mrow + r) * OUT_SZ + ncol] = acc[mi][ni][r];
    }
  }
}

// ---------------------------------------------------------------------------
// Kernel 4: LayerNorm (over OUT) + affine + PReLU, one block per row
// ---------------------------------------------------------------------------
__global__ __launch_bounds__(256) void kan_ln_prelu(
    const float* __restrict__ Y, const float* __restrict__ gamma,
    const float* __restrict__ beta, const float* __restrict__ pa,
    float* __restrict__ out)
{
  const int row = blockIdx.x;
  const float* y = Y + (size_t)row * OUT_SZ;

  float v[4]; float s = 0.0f, s2 = 0.0f;
#pragma unroll
  for (int j = 0; j < 4; ++j) {
    int c = threadIdx.x + j * 256;
    v[j] = y[c]; s += v[j]; s2 += v[j] * v[j];
  }
#pragma unroll
  for (int o = 16; o; o >>= 1) {
    s  += __shfl_xor(s, o, 32);
    s2 += __shfl_xor(s2, o, 32);
  }

  __shared__ float rs[8], rs2[8], stat[2];
  int lane = threadIdx.x & 31, wave = threadIdx.x >> 5;
  if (lane == 0) { rs[wave] = s; rs2[wave] = s2; }
  __syncthreads();
  if (threadIdx.x == 0) {
    float ts = 0.0f, ts2 = 0.0f;
    for (int w = 0; w < 8; ++w) { ts += rs[w]; ts2 += rs2[w]; }
    float mu  = ts / (float)OUT_SZ;
    float var = ts2 / (float)OUT_SZ - mu * mu;
    stat[0] = mu; stat[1] = rsqrtf(var + 1e-5f);
  }
  __syncthreads();

  float mu = stat[0], rinv = stat[1];
  float a = pa[0];
#pragma unroll
  for (int j = 0; j < 4; ++j) {
    int c = threadIdx.x + j * 256;
    float yn = (v[j] - mu) * rinv * gamma[c] + beta[c];
    out[(size_t)row * OUT_SZ + c] = (yn >= 0.0f) ? yn : a * yn;
  }
}

// ---------------------------------------------------------------------------
extern "C" void kernel_launch(void* const* d_in, const int* in_sizes, int n_in,
                              void* d_out, int out_size, void* d_ws, size_t ws_size,
                              hipStream_t stream)
{
  const float* x     = (const float*)d_in[0];
  const float* bw    = (const float*)d_in[1];
  const float* sw    = (const float*)d_in[2];
  const float* grid  = (const float*)d_in[3];
  const float* gamma = (const float*)d_in[4];
  const float* beta  = (const float*)d_in[5];
  const float* pa    = (const float*)d_in[6];

  char* ws = (char*)d_ws;
  half_t* Ag  = (half_t*)ws;  ws += (size_t)B_SZ  * K1 * sizeof(half_t);  //  16 MB
  half_t* As  = (half_t*)ws;  ws += (size_t)B_SZ  * K2 * sizeof(half_t);  // 128 MB
  half_t* Wbh = (half_t*)ws;  ws += (size_t)OUT_SZ * K1 * sizeof(half_t); //   2 MB
  half_t* Wsh = (half_t*)ws;  ws += (size_t)OUT_SZ * K2 * sizeof(half_t); //  16 MB
  float*  Yb  = (float*)ws;                                               //  32 MB

  kan_prep<<<(B_SZ * IN_SZ) / 256, 256, 0, stream>>>(x, grid, Ag, As);
  kan_pack<<<(OUT_SZ * IN_SZ * (1 + KSPL)) / 256, 256, 0, stream>>>(bw, sw, Wbh, Wsh);

  dim3 gg(OUT_SZ / BLK_N, B_SZ / BLK_M);   // 8 x 64 blocks
  kan_wmma_gemm<<<gg, 256, 0, stream>>>(Ag, As, Wbh, Wsh, Yb);

  kan_ln_prelu<<<B_SZ, 256, 0, stream>>>(Yb, gamma, beta, pa, (float*)d_out);
}
